// RelativeHead_25606595019371
// MI455X (gfx1250) — matrix operand order repaired and checked by reading, not compiled, not run on hardware
//
#include <hip/hip_runtime.h>
#include <hip/hip_bf16.h>

#ifndef __has_builtin
#define __has_builtin(x) 0
#endif

#if __has_builtin(__builtin_amdgcn_global_load_async_to_lds_b128)
#define HAVE_ASYNC_LDS 1
#else
#define HAVE_ASYNC_LDS 0
#endif

#define AS1 __attribute__((address_space(1)))
#define AS3 __attribute__((address_space(3)))

typedef __bf16 bf16_t;
typedef __attribute__((ext_vector_type(16))) __bf16 v16bf;
typedef __attribute__((ext_vector_type(8)))  __bf16 v8bf;
typedef __attribute__((ext_vector_type(8)))  float  v8f;
typedef __attribute__((ext_vector_type(4)))  int    v4i;

union FragAB { v16bf v; v8bf h[2]; };
union FragC  { v8f  v; float e[8]; };
union Pack8  { v8bf v; bf16_t e[8]; };

// ---- WMMA wrapper: D = A(16x32 bf16) x B(32x16 bf16) + C(f32) ----
__device__ inline v8f wmma_bf16(v16bf a, v16bf b, v8f c) {
  return __builtin_amdgcn_wmma_f32_16x16x32_bf16(
      /*neg_a=*/false, a, /*neg_b=*/false, b,
      /*c_mod=*/(short)0, c, /*reuse_a=*/false, /*reuse_b=*/false);
}

#if HAVE_ASYNC_LDS
// 16B async copy global -> LDS (tracked by ASYNCcnt)
__device__ inline void async_copy16(const void* g, void* l) {
  void* gv = const_cast<void*>(g);
  __builtin_amdgcn_global_load_async_to_lds_b128(
      (AS1 v4i*)gv, (AS3 v4i*)l, /*offset=*/0, /*cpol=*/0);
}
#endif

// A-fragment (16x32, MxK) from row-major bf16 source, tile at (row0,k0).
// ISA layout: lane m=L&15 holds row M=m, K in {kb..kb+7, kb+16..kb+23}, kb=(L>>4)*8.
__device__ inline v16bf load_frag_a_bf16(const bf16_t* src, int ld, int row0, int k0, int lane) {
  int m  = lane & 15;
  int kb = (lane >> 4) << 3;
  const bf16_t* p = src + (size_t)(row0 + m) * ld + k0 + kb;
  FragAB f;
  f.h[0] = *(const v8bf*)(p);        // K = kb .. kb+7
  f.h[1] = *(const v8bf*)(p + 16);   // K = kb+16 .. kb+23
  return f.v;
}

// A-fragment from f32 source, converting to bf16 in-register (x is read only here).
__device__ inline v16bf load_frag_a_f32(const float* src, int ld, int row0, int k0, int lane) {
  int m  = lane & 15;
  int kb = (lane >> 4) << 3;
  const float* p = src + (size_t)(row0 + m) * ld + k0 + kb;
  v8f lo = *(const v8f*)(p);
  v8f hi = *(const v8f*)(p + 16);
  FragAB f;
#pragma unroll
  for (int i = 0; i < 8; ++i) { f.h[0][i] = (bf16_t)lo[i]; f.h[1][i] = (bf16_t)hi[i]; }
  return f.v;
}

// B-fragment (32x16, KxN) loaded from B^T stored row-major [n][k] (contiguous per lane).
// ISA layout: lane n=L&15 holds col N=n, K in {kb..kb+7, kb+16..kb+23}, kb=(L>>4)*8.
__device__ inline v16bf load_frag_bT(const bf16_t* srcT, int ld, int k0, int n0, int lane) {
  int n  = lane & 15;
  int kb = (lane >> 4) << 3;
  const bf16_t* p = srcT + (size_t)(n0 + n) * ld + k0 + kb;
  FragAB f;
  f.h[0] = *(const v8bf*)(p);
  f.h[1] = *(const v8bf*)(p + 16);
  return f.v;
}

// ---- Kernel 0: convert weights to bf16, transposed; R[0:256] to bf16 ----
__global__ void convert_kernel(const float* __restrict__ Wk, const float* __restrict__ Wq,
                               const float* __restrict__ Wv, const float* __restrict__ R,
                               bf16_t* __restrict__ WqT, bf16_t* __restrict__ WkT,
                               bf16_t* __restrict__ WvT, bf16_t* __restrict__ Rh) {
  int tid = blockIdx.x * blockDim.x + threadIdx.x;
  if (tid < 3 * 16384) {
    int a = tid / 16384, rem = tid % 16384;
    int h = rem >> 8, c = rem & 255;                 // WT[h][c] = W[c][h]
    const float* W = (a == 0) ? Wq : (a == 1) ? Wk : Wv;
    bf16_t* WT     = (a == 0) ? WqT : (a == 1) ? WkT : WvT;
    WT[h * 256 + c] = (bf16_t)W[c * 64 + h];
  } else {
    int rem = tid - 3 * 16384;
    if (rem < 16384) Rh[rem] = (bf16_t)R[rem];       // first 256 rows of rel_pos_embedding
  }
}

// ---- Kernel 1: QKV projection. q,k stored bf16 [b*T+t][64]; v stored transposed [b][64][256]
__global__ __launch_bounds__(128) void qkv_kernel(
    const float* __restrict__ x,
    const bf16_t* __restrict__ WqT, const bf16_t* __restrict__ WkT, const bf16_t* __restrict__ WvT,
    bf16_t* __restrict__ qh, bf16_t* __restrict__ kh, bf16_t* __restrict__ vT) {
  const int lane = threadIdx.x & 31;
  const int wave = threadIdx.x >> 5;      // 4 waves: one 16-col n-tile each
  const int row0 = blockIdx.x * 16;       // 16-row tile of the 65536 rows
  const int n0   = wave * 16;

  v8f accq = {}, acck = {}, accv = {};
  // software pipeline: fragments for step kk+1 are in flight while step kk runs on the matrix pipe
  v16bf a  = load_frag_a_f32(x, 256, row0, 0, lane);
  v16bf bq = load_frag_bT(WqT, 256, 0, n0, lane);
  v16bf bk = load_frag_bT(WkT, 256, 0, n0, lane);
  v16bf bv = load_frag_bT(WvT, 256, 0, n0, lane);
#pragma unroll
  for (int kk = 0; kk < 8; ++kk) {        // K = 256 in steps of 32
    v16bf a_n = a, bq_n = bq, bk_n = bk, bv_n = bv;
    if (kk < 7) {
      int k0 = (kk + 1) * 32;
      __builtin_prefetch(x + (size_t)(row0 + (lane & 15)) * 256 + k0 + 32, 0, 0);
      a_n  = load_frag_a_f32(x, 256, row0, k0, lane);
      bq_n = load_frag_bT(WqT, 256, k0, n0, lane);
      bk_n = load_frag_bT(WkT, 256, k0, n0, lane);
      bv_n = load_frag_bT(WvT, 256, k0, n0, lane);
    }
    accq = wmma_bf16(a, bq, accq);
    acck = wmma_bf16(a, bk, acck);
    accv = wmma_bf16(a, bv, accv);
    a = a_n; bq = bq_n; bk = bk_n; bv = bv_n;
  }

  const int n = lane & 15, hi = lane >> 4;
  FragC cq, ck, cv; cq.v = accq; ck.v = acck; cv.v = accv;
#pragma unroll
  for (int r = 0; r < 8; ++r) {           // C/D layout: row M = r + 8*hi, col N = n
    int row = row0 + r + 8 * hi;
    qh[(size_t)row * 64 + n0 + n] = (bf16_t)cq.e[r];
    kh[(size_t)row * 64 + n0 + n] = (bf16_t)ck.e[r];
  }
  // v transposed: vT[b][h][t] -> contiguous 8 bf16 per lane along t
  int b    = row0 >> 8;
  int tloc = (row0 & 255) + 8 * hi;
  Pack8 st;
#pragma unroll
  for (int r = 0; r < 8; ++r) st.e[r] = (bf16_t)cv.e[r];
  *(v8bf*)(vT + (size_t)b * 64 * 256 + (size_t)(n0 + n) * 256 + tloc) = st.v;
}

// ---- Kernel 2: fused attention per (batch, 16-row t-tile) ----
// S[t,s] = 0.125*(q.k) + q.R[s-t+255]; causal mask; softmax; out = S @ v
// Dynamic LDS layout (74752 B):
//   [0)      Ks   bf16 [256][64]  32768  (async-staged K for this batch)
//   [32768)  P    f32  [16][256]  16384
//   [49152)  S    f32  [16][256]  16384
//   [65536)  Wp   bf16 [16][256]   8192
//   [73728)  red  f32  [16][16]    1024
#define ATTN_LDS_BYTES 74752
__global__ __launch_bounds__(256) void attn_kernel(
    const bf16_t* __restrict__ qh, const bf16_t* __restrict__ kh,
    const bf16_t* __restrict__ vT, const bf16_t* __restrict__ Rh,
    float* __restrict__ out) {
  extern __shared__ char smem[];
  bf16_t* Ks       = (bf16_t*)smem;
  float (*P)[256]  = (float(*)[256])(smem + 32768);
  float (*S)[256]  = (float(*)[256])(smem + 49152);
  bf16_t* Wp       = (bf16_t*)(smem + 65536);
  float* red       = (float*)(smem + 73728);

  const int lane  = threadIdx.x & 31;
  const int wave  = threadIdx.x >> 5;    // 8 waves
  const int ttile = blockIdx.x;          // 0..15
  const int bb    = blockIdx.y;          // batch
  const int t0    = ttile * 16;
  const bf16_t* qb = qh + (size_t)bb * 256 * 64;
  const bf16_t* kb = kh + (size_t)bb * 256 * 64;
  const bf16_t* vb = vT + (size_t)bb * 64 * 256;

  // Kick off async copy of this batch's K (32KB) into LDS; overlaps with Step A.
  {
    const char* src = (const char*)kb;
#pragma unroll
    for (int i = 0; i < 8; ++i) {
      int off = ((int)threadIdx.x + i * 256) * 16;
#if HAVE_ASYNC_LDS
      async_copy16(src + off, (char*)smem + off);
#else
      *(v8bf*)((char*)smem + off) = *(const v8bf*)(src + off);
#endif
    }
  }

  // q-tile A-fragments: shared by Steps A and B, load once.
  const v16bf aq0 = load_frag_a_bf16(qb, 64, t0, 0, lane);
  const v16bf aq1 = load_frag_a_bf16(qb, 64, t0, 32, lane);

  // Step A: P = q_tile @ R^T over all 256 d-columns (causal => d in [0,255])
  for (int dt = wave; dt < 16; dt += 8) {
    v8f acc = {};
    acc = wmma_bf16(aq0, load_frag_bT(Rh, 64, 0, dt * 16, lane), acc);
    acc = wmma_bf16(aq1, load_frag_bT(Rh, 64, 32, dt * 16, lane), acc);
    int n = lane & 15, hi = lane >> 4;
    FragC c; c.v = acc;
#pragma unroll
    for (int r = 0; r < 8; ++r) P[r + 8 * hi][dt * 16 + n] = c.e[r];
  }
  // init S to -inf (masked / uncomputed tiles)
  {
    float* s = &S[0][0];
    for (int i = threadIdx.x; i < 16 * 256; i += 256) s[i] = -__builtin_inff();
  }
#if HAVE_ASYNC_LDS
#if __has_builtin(__builtin_amdgcn_s_wait_asynccnt)
  __builtin_amdgcn_s_wait_asynccnt(0);
#else
  asm volatile("s_wait_asynccnt 0" ::: "memory");
#endif
#endif
  __syncthreads();   // P + S-init + per-wave async K copies all complete

  // Step B: content scores for s-tiles 0..ttile from LDS K, fused position gather + mask
  for (int st = wave; st <= ttile; st += 8) {
    v8f acc = {};
    acc = wmma_bf16(aq0, load_frag_bT(Ks, 64, 0, st * 16, lane), acc);
    acc = wmma_bf16(aq1, load_frag_bT(Ks, 64, 32, st * 16, lane), acc);
    int n = lane & 15, hi = lane >> 4;
    int s_g = st * 16 + n;
    FragC c; c.v = acc;
#pragma unroll
    for (int r = 0; r < 8; ++r) {
      int m   = r + 8 * hi;
      int t_g = t0 + m;
      if (s_g <= t_g) {
        int d = s_g - t_g + 255;         // in [0,255]
        S[m][s_g] = c.e[r] * 0.125f + P[m][d];
      }
    }
  }
  __syncthreads();

  // Step C: softmax over 256 columns; 16 threads per row, 16 cols per thread
  {
    int row = threadIdx.x >> 4;
    int seg = threadIdx.x & 15;
    float lm = -__builtin_inff();
#pragma unroll
    for (int j = 0; j < 16; ++j) lm = fmaxf(lm, S[row][seg * 16 + j]);
    red[row * 16 + seg] = lm;
    __syncthreads();
    float m = -__builtin_inff();
#pragma unroll
    for (int j = 0; j < 16; ++j) m = fmaxf(m, red[row * 16 + j]);
    __syncthreads();
    float lsum = 0.f;
#pragma unroll
    for (int j = 0; j < 16; ++j) {
      float e = __expf(S[row][seg * 16 + j] - m);  // exp(-inf - m) = 0
      S[row][seg * 16 + j] = e;
      lsum += e;
    }
    red[row * 16 + seg] = lsum;
    __syncthreads();
    float sum = 0.f;
#pragma unroll
    for (int j = 0; j < 16; ++j) sum += red[row * 16 + j];
    float inv = 1.0f / sum;
#pragma unroll
    for (int j = 0; j < 16; ++j)
      Wp[row * 256 + seg * 16 + j] = (bf16_t)(S[row][seg * 16 + j] * inv);
  }
  __syncthreads();

  // Step D: out_tile = Wp(16x256) @ v(256x64); waves 0..3 take one 16-col n-tile each
  if (wave < 4) {
    int n0 = wave * 16;
    v8f acc = {};
#pragma unroll
    for (int kk = 0; kk < 8; ++kk) {
      v16bf a = load_frag_a_bf16(Wp, 256, 0, kk * 32, lane);   // LDS reads
      v16bf B = load_frag_bT(vb, 256, kk * 32, n0, lane);      // vT row-major
      acc = wmma_bf16(a, B, acc);
    }
    int n = lane & 15, hi = lane >> 4;
    FragC c; c.v = acc;
#pragma unroll
    for (int r = 0; r < 8; ++r) {
      int m = r + 8 * hi;
      out[((size_t)bb * 256 + t0 + m) * 64 + n0 + n] = c.e[r];
    }
  }
}

extern "C" void kernel_launch(void* const* d_in, const int* in_sizes, int n_in,
                              void* d_out, int out_size, void* d_ws, size_t ws_size,
                              hipStream_t stream) {
  const float* x  = (const float*)d_in[0];
  const float* Wk = (const float*)d_in[1];
  const float* Wq = (const float*)d_in[2];
  const float* Wv = (const float*)d_in[3];
  const float* R  = (const float*)d_in[4];
  float* out = (float*)d_out;

  char* ws = (char*)d_ws;
  // Workspace layout (~24.1 MB used)
  bf16_t* qh  = (bf16_t*)(ws);                        // 65536*64 bf16 = 8 MiB
  bf16_t* kh  = (bf16_t*)(ws + (size_t)8 * 1024 * 1024);
  bf16_t* vT  = (bf16_t*)(ws + (size_t)16 * 1024 * 1024);
  bf16_t* WqT = (bf16_t*)(ws + (size_t)24 * 1024 * 1024);
  bf16_t* WkT = (bf16_t*)(ws + (size_t)24 * 1024 * 1024 + 32768);
  bf16_t* WvT = (bf16_t*)(ws + (size_t)24 * 1024 * 1024 + 65536);
  bf16_t* Rh  = (bf16_t*)(ws + (size_t)24 * 1024 * 1024 + 98304);

  convert_kernel<<<256, 256, 0, stream>>>(Wk, Wq, Wv, R, WqT, WkT, WvT, Rh);
  qkv_kernel<<<4096, 128, 0, stream>>>(x, WqT, WkT, WvT, qh, kh, vT);
  dim3 grid(16, 256);
  attn_kernel<<<grid, 256, ATTN_LDS_BYTES, stream>>>(qh, kh, vT, Rh, out);
}
